// GraphPARC_1769526526738
// MI455X (gfx1250) — matrix operand order repaired and checked by reading, not compiled, not run on hardware
//
#include <hip/hip_runtime.h>

typedef __attribute__((ext_vector_type(16))) _Float16 v16h;
typedef __attribute__((ext_vector_type(8)))  float    v8f;

#define DEG 8
#define NTIME 2

// sigmoid via native v_exp_f32 + single v_rcp_f32 (no IEEE div_scale/NR chain).
__device__ __forceinline__ float sigmoidf_(float x) {
  return __builtin_amdgcn_rcpf(1.0f + __expf(-x));
}
__device__ __forceinline__ float tanhf_(float x) {
  return 2.0f * sigmoidf_(2.0f * x) - 1.0f;
}

// A operand: 16x32 f16 tile, row-major in LDS (row stride = `stride` halves).
// Lane mapping (contiguous K, matching B fragments): lanes 0-15 -> M rows, K 0..15;
// lanes 16-31 -> same rows, K 16..31. Per-lane runs are contiguous ->
// compiler emits 2x ds_load_b128.
__device__ __forceinline__ v16h load_tileA(const _Float16* __restrict__ base, int stride, int lane) {
  const int half = lane >> 4;
  const int m    = lane & 15;
  const _Float16* row = base + m * stride;
  v16h a;
#pragma unroll
  for (int d = 0; d < 8; ++d) {
    int k = d * 2 + half * 16;
    a[2 * d]     = row[k];
    a[2 * d + 1] = row[k + 1];
  }
  return a;
}

// B operand pre-swizzled into fragment layout at staging time:
// frag[( tile*32 + lane )*16 + e] = B[k0 + (lane>>4)*16 + e][n0 + (lane&15)].
// Consuming it is one aligned v16h LDS load (2x ds_load_b128) per tile.
__device__ __forceinline__ v16h load_fragB(const _Float16* __restrict__ base, int tile, int lane) {
  return *(const v16h*)(base + ((size_t)tile * 32 + lane) * 16);
}

template <int CIN, int COUT>
struct LayerDims {
  static constexpr int G     = 4 * CIN;
  static constexpr int GP    = ((G + 15) / 16) * 16;      // gate cols padded to 16
  static constexpr int CINP  = (CIN <= 32) ? 32 : 64;     // K padded to WMMA K=32 multiple
  static constexpr int COUTP = ((COUT + 15) / 16) * 16;
  static constexpr int KT    = CINP / 32;                 // K tiles (1 or 2)
  static constexpr int GT    = GP / 16;                   // gate column tiles
  static constexpr int CT    = COUTP / 16;                // output column tiles
  // dynamic shared memory layout (byte offsets; every block is 32B-aligned)
  static constexpr size_t oG   = 0;                                      // 16*GP f32
  static constexpr size_t oCC  = oG   + sizeof(float) * 16 * GP;         // 16*CIN f32
  static constexpr size_t oB   = oCC  + sizeof(float) * 16 * CIN;        // GP f32
  static constexpr size_t oBL  = oB   + sizeof(float) * GP;              // COUTP f32
  static constexpr size_t oWih = oBL  + sizeof(float) * COUTP;           // KT*GT*512 f16 (swizzled)
  static constexpr size_t oWhh = oWih + sizeof(_Float16) * KT * GT * 512;
  static constexpr size_t oWl  = oWhh + sizeof(_Float16) * KT * GT * 512; // KT*CT*512 f16 (swizzled)
  static constexpr size_t oWr  = oWl  + sizeof(_Float16) * KT * CT * 512;
  static constexpr size_t oXj  = oWr  + sizeof(_Float16) * KT * CT * 512; // DEG*16*CINP f16
  static constexpr size_t oXn  = oXj  + sizeof(_Float16) * DEG * 16 * CINP;
  static constexpr size_t oH   = oXn  + sizeof(_Float16) * 16 * CINP;
  static constexpr size_t BYTES = oH + sizeof(_Float16) * 16 * CINP;
};

// One workgroup = 16 nodes (M tile), 4 waves split the gate-column tiles.
// Fuses: neighbor gather -> 8-step LSTM (two WMMA GEMM chains per step) ->
//        output projection (h@Wl + x@Wr) -> bias (+ReLU) -> store.
template <int CIN, int COUT>
__global__ __launch_bounds__(128)
void sage_lstm_layer(const float* __restrict__ x, float* __restrict__ y,
                     const int* __restrict__ src,
                     const float* __restrict__ Wih, const float* __restrict__ Whh,
                     const float* __restrict__ bG,
                     const float* __restrict__ Wl,  const float* __restrict__ blG,
                     const float* __restrict__ Wr,
                     int Ntot, int relu) {
  using D = LayerDims<CIN, COUT>;
  extern __shared__ char smem[];
  float*    gF   = (float*)(smem + D::oG);
  float*    ccF  = (float*)(smem + D::oCC);
  float*    bF   = (float*)(smem + D::oB);
  float*    blF  = (float*)(smem + D::oBL);
  _Float16* wihS = (_Float16*)(smem + D::oWih);
  _Float16* whhS = (_Float16*)(smem + D::oWhh);
  _Float16* wlS  = (_Float16*)(smem + D::oWl);
  _Float16* wrS  = (_Float16*)(smem + D::oWr);
  _Float16* xjH  = (_Float16*)(smem + D::oXj);
  _Float16* xnH  = (_Float16*)(smem + D::oXn);
  _Float16* hH   = (_Float16*)(smem + D::oH);

  const int tid  = threadIdx.x;
  const int lane = tid & 31;
  const int wave = tid >> 5;
  const int gnb  = blockIdx.x * 16;

  // ---- stage gate weights: f32 -> f16, zero-padded, pre-swizzled to B-fragment layout ----
  for (int idx = tid; idx < D::KT * D::GT * 512; idx += 128) {
    int tile = idx >> 9;            // fragment tile (kt*GT + j)
    int r    = idx & 511;
    int l    = r >> 4;              // lane within fragment
    int e    = r & 15;              // element within lane
    int kt   = tile / D::GT;
    int j    = tile - kt * D::GT;
    int k    = kt * 32 + (l >> 4) * 16 + e;   // source K row
    int c    = j * 16 + (l & 15);             // source N col
    bool ok  = (k < CIN) && (c < D::G);
    wihS[idx] = (_Float16)(ok ? Wih[k * D::G + c] : 0.0f);
    whhS[idx] = (_Float16)(ok ? Whh[k * D::G + c] : 0.0f);
  }
  // ---- stage projection weights likewise ----
  for (int idx = tid; idx < D::KT * D::CT * 512; idx += 128) {
    int tile = idx >> 9;
    int r    = idx & 511;
    int l    = r >> 4;
    int e    = r & 15;
    int kt   = tile / D::CT;
    int j    = tile - kt * D::CT;
    int k    = kt * 32 + (l >> 4) * 16 + e;
    int c    = j * 16 + (l & 15);
    bool ok  = (k < CIN) && (c < COUT);
    wlS[idx] = (_Float16)(ok ? Wl[k * COUT + c] : 0.0f);
    wrS[idx] = (_Float16)(ok ? Wr[k * COUT + c] : 0.0f);
  }
  for (int idx = tid; idx < D::GP; idx += 128)
    bF[idx] = (idx < D::G) ? bG[idx] : 0.0f;
  for (int idx = tid; idx < D::COUTP; idx += 128)
    blF[idx] = (idx < COUT) ? blG[idx] : 0.0f;

  // ---- gather neighbor features (L2-resident random gather) ----
  for (int idx = tid; idx < DEG * 16 * D::CINP; idx += 128) {
    int t = idx / (16 * D::CINP);
    int r = idx - t * (16 * D::CINP);
    int m = r / D::CINP, k = r - m * D::CINP;
    int gn = gnb + m;
    int gc = (gn < Ntot) ? gn : 0;
    int nb = src[gc * DEG + t];
    xjH[(t * 16 + m) * D::CINP + k] = (_Float16)((k < CIN) ? x[(size_t)nb * CIN + k] : 0.0f);
  }
  for (int idx = tid; idx < 16 * D::CINP; idx += 128) {
    int m = idx / D::CINP, k = idx - m * D::CINP;
    int gn = gnb + m;
    int gc = (gn < Ntot) ? gn : 0;
    xnH[idx] = (_Float16)((k < CIN) ? x[(size_t)gc * CIN + k] : 0.0f);
    hH[idx]  = (_Float16)0.0f;
  }
  for (int idx = tid; idx < 16 * CIN; idx += 128) ccF[idx] = 0.0f;
  __syncthreads();

  // ---- LSTM over DEG neighbor steps ----
  for (int t = 0; t < DEG; ++t) {
    // A fragments are invariant over gate-column tiles: load once per step.
    v16h aX[D::KT], aH[D::KT];
#pragma unroll
    for (int kt = 0; kt < D::KT; ++kt) {
      aX[kt] = load_tileA(xjH + (t * 16) * D::CINP + kt * 32, D::CINP, lane);
      aH[kt] = load_tileA(hH + kt * 32, D::CINP, lane);
    }
    for (int j = wave; j < D::GT; j += 4) {
      v8f acc = {};
#pragma unroll
      for (int kt = 0; kt < D::KT; ++kt) {
        acc = __builtin_amdgcn_wmma_f32_16x16x32_f16(false, aX[kt], false,
                  load_fragB(wihS, kt * D::GT + j, lane), (short)0, acc, false, false);
        acc = __builtin_amdgcn_wmma_f32_16x16x32_f16(false, aH[kt], false,
                  load_fragB(whhS, kt * D::GT + j, lane), (short)0, acc, false, false);
      }
      const int half = lane >> 4;
      const int n = (lane & 15) + j * 16;
#pragma unroll
      for (int v = 0; v < 8; ++v) gF[(v + 8 * half) * D::GP + n] = acc[v];
    }
    __syncthreads();
    // gate nonlinearities (native v_exp_f32 / v_rcp_f32 TRANS ops)
    for (int idx = tid; idx < 16 * CIN; idx += 128) {
      int m = idx / CIN, c = idx - m * CIN;
      const float* gr = gF + m * D::GP;
      float gi = gr[c]           + bF[c];
      float gf = gr[CIN + c]     + bF[CIN + c];
      float gg = gr[2 * CIN + c] + bF[2 * CIN + c];
      float go = gr[3 * CIN + c] + bF[3 * CIN + c];
      float cc = sigmoidf_(gf) * ccF[idx] + sigmoidf_(gi) * tanhf_(gg);
      ccF[idx] = cc;
      hH[m * D::CINP + c] = (_Float16)(sigmoidf_(go) * tanhf_(cc));
    }
    __syncthreads();
  }

  // ---- output projection: h @ Wl + x @ Wr ----
  {
    v16h aH[D::KT], aN[D::KT];
#pragma unroll
    for (int kt = 0; kt < D::KT; ++kt) {
      aH[kt] = load_tileA(hH + kt * 32, D::CINP, lane);
      aN[kt] = load_tileA(xnH + kt * 32, D::CINP, lane);
    }
    for (int j = wave; j < D::CT; j += 4) {
      v8f acc = {};
#pragma unroll
      for (int kt = 0; kt < D::KT; ++kt) {
        acc = __builtin_amdgcn_wmma_f32_16x16x32_f16(false, aH[kt], false,
                  load_fragB(wlS, kt * D::CT + j, lane), (short)0, acc, false, false);
        acc = __builtin_amdgcn_wmma_f32_16x16x32_f16(false, aN[kt], false,
                  load_fragB(wrS, kt * D::CT + j, lane), (short)0, acc, false, false);
      }
      const int half = lane >> 4;
      const int n = (lane & 15) + j * 16;
#pragma unroll
      for (int v = 0; v < 8; ++v) gF[(v + 8 * half) * D::GP + n] = acc[v];
    }
  }
  __syncthreads();
  for (int idx = tid; idx < 16 * COUT; idx += 128) {
    int m = idx / COUT, c = idx - m * COUT;
    int gn = gnb + m;
    float v = gF[m * D::GP + c] + blF[c];
    if (relu) v = fmaxf(v, 0.0f);
    if (gn < Ntot) y[(size_t)gn * COUT + c] = v;
  }
}

// ---- small elementwise helpers ----
__global__ void k_init_F(float* __restrict__ F, const float* __restrict__ pressure, int N) {
  int i = blockIdx.x * blockDim.x + threadIdx.x;
  if (i < N) F[i] = pressure[i * 2];
}
__global__ void k_concat_diff(float* __restrict__ out, const float* __restrict__ F,
                              const float* __restrict__ attr, int N) {
  int i = blockIdx.x * blockDim.x + threadIdx.x;
  if (i < N) {
    out[i * 8] = F[i];
#pragma unroll
    for (int c = 0; c < 7; ++c) out[i * 8 + 1 + c] = attr[i * 7 + c];
  }
}
__global__ void k_concat_integ(float* __restrict__ out, const float* __restrict__ F,
                               const float* __restrict__ Fdot, int N) {
  int i = blockIdx.x * blockDim.x + threadIdx.x;
  if (i < N) { out[i * 2] = F[i]; out[i * 2 + 1] = Fdot[i]; }
}
__global__ void k_update(float* __restrict__ F, const float* __restrict__ delta,
                         const float* __restrict__ Fdot,
                         float* __restrict__ outFs, float* __restrict__ outFd, int N, int t) {
  int i = blockIdx.x * blockDim.x + threadIdx.x;
  if (i < N) {
    float f = F[i] + delta[i];
    F[i] = f;
    outFs[i * NTIME + t] = f;
    outFd[i * NTIME + t] = Fdot[i];
  }
}

// Param layout: jax tree-flatten of params dict -> 'diff' layers 0..6 then 'integ' 0..6,
// each conv dict by sorted key: Whh, Wih, Wl, Wr, b, bl  (6 arrays per layer).
template <int CIN, int COUT>
static void run_layer(const float* x, float* y, const int* src,
                      void* const* P, int layer, int relu, int N, hipStream_t s) {
  const float* Whh = (const float*)P[layer * 6 + 0];
  const float* Wih = (const float*)P[layer * 6 + 1];
  const float* Wl  = (const float*)P[layer * 6 + 2];
  const float* Wr  = (const float*)P[layer * 6 + 3];
  const float* b   = (const float*)P[layer * 6 + 4];
  const float* bl  = (const float*)P[layer * 6 + 5];
  dim3 grid((N + 15) / 16);
  sage_lstm_layer<CIN, COUT><<<grid, 128, (unsigned)LayerDims<CIN, COUT>::BYTES, s>>>(
      x, y, src, Wih, Whh, b, Wl, bl, Wr, N, relu);
}

extern "C" void kernel_launch(void* const* d_in, const int* in_sizes, int n_in,
                              void* d_out, int out_size, void* d_ws, size_t ws_size,
                              hipStream_t stream) {
  (void)out_size; (void)ws_size;
  const int N = in_sizes[0] / 2;                    // pressure is (N,2)
  const float* pressure  = (const float*)d_in[0];
  const float* node_attr = (const float*)d_in[1];
  void* const* P = d_in + 2;                        // 84 param arrays
  const int* src = (const int*)d_in[n_in - 2];      // edge_index row 0 (src)

  float* outFs = (float*)d_out;                     // (N, NT, 1)
  float* outFd = outFs + (size_t)N * NTIME;         // (N, NT, 1)

  float* bufA = (float*)d_ws;                       // N*64 activations (ping)
  float* bufB = bufA + (size_t)N * 64;              // N*64 activations (pong)
  float* F    = bufB + (size_t)N * 64;              // N
  float* Fdot = F + N;                              // N

  dim3 eb(256), eg((N + 255) / 256);
  k_init_F<<<eg, eb, 0, stream>>>(F, pressure, N);

  for (int t = 0; t < NTIME; ++t) {
    // Fdot = diff_net([F, node_attr])
    k_concat_diff<<<eg, eb, 0, stream>>>(bufA, F, node_attr, N);
    run_layer<8, 16>(bufA, bufB, src, P, 0, 1, N, stream);
    run_layer<16, 32>(bufB, bufA, src, P, 1, 1, N, stream);
    run_layer<32, 64>(bufA, bufB, src, P, 2, 1, N, stream);
    run_layer<64, 64>(bufB, bufA, src, P, 3, 1, N, stream);
    run_layer<64, 32>(bufA, bufB, src, P, 4, 1, N, stream);
    run_layer<32, 16>(bufB, bufA, src, P, 5, 1, N, stream);
    run_layer<16, 1>(bufA, Fdot, src, P, 6, 0, N, stream);

    // F += integ_net([F, Fdot])
    k_concat_integ<<<eg, eb, 0, stream>>>(bufA, F, Fdot, N);
    run_layer<2, 16>(bufA, bufB, src, P, 7, 1, N, stream);
    run_layer<16, 32>(bufB, bufA, src, P, 8, 1, N, stream);
    run_layer<32, 64>(bufA, bufB, src, P, 9, 1, N, stream);
    run_layer<64, 64>(bufB, bufA, src, P, 10, 1, N, stream);
    run_layer<64, 32>(bufA, bufB, src, P, 11, 1, N, stream);
    run_layer<32, 16>(bufB, bufA, src, P, 12, 1, N, stream);
    run_layer<16, 1>(bufA, bufB, src, P, 13, 0, N, stream);  // delta (stride 1) in bufB

    k_update<<<eg, eb, 0, stream>>>(F, bufB, Fdot, outFs, outFd, N, t);
  }
}
